// Net2_45930380263896
// MI455X (gfx1250) — matrix operand order repaired and checked by reading, not compiled, not run on hardware
//
#include <hip/hip_runtime.h>
#include <hip/hip_bf16.h>

// ---------------------------------------------------------------------------
// Problem constants (from reference)
// ---------------------------------------------------------------------------
#define QH   64     // hidden size
#define DD   64     // input feature size
#define CTR  64     // CTRAJ
#define NN   32     // N
#define TT   256    // TX == TY
#define PP   8
#define LL   512
#define G4   256    // 4*Q

typedef __attribute__((ext_vector_type(16))) _Float16 v16h;
typedef __attribute__((ext_vector_type(8)))  _Float16 v8h;
typedef __attribute__((ext_vector_type(8)))  float    v8f;

// ---- optional gfx1250 fast paths (guarded so compile can never break) ------
#if defined(__has_builtin)
#  if __has_builtin(__builtin_amdgcn_global_load_async_to_lds_b128)
#    define HAS_ASYNC_LDS 1
#  endif
#  if __has_builtin(__builtin_amdgcn_tanhf)
#    define fast_tanhf __builtin_amdgcn_tanhf
#  elif __has_builtin(__builtin_amdgcn_tanh_f32)
#    define fast_tanhf __builtin_amdgcn_tanh_f32
#  endif
#endif
#ifndef fast_tanhf
#  define fast_tanhf tanhf
#endif

#ifdef HAS_ASYNC_LDS
// Parameter types per hipcc diagnostic: <4 x i32> pointers, addrspace(1) src
// ("__device__"), addrspace(3) dst ("__shared__").
typedef __attribute__((vector_size(16))) int i128_t;
typedef __attribute__((address_space(1))) i128_t as1_i128;
typedef __attribute__((address_space(3))) i128_t as3_i128;
__device__ __forceinline__ void async_b128(const void* g, void* l) {
    __builtin_amdgcn_global_load_async_to_lds_b128(
        (as1_i128*)(unsigned long long)g,
        (as3_i128*)(unsigned long long)l, 0, 0);
}
#  if __has_builtin(__builtin_amdgcn_s_wait_asynccnt)
#    define WAIT_ASYNC() __builtin_amdgcn_s_wait_asynccnt(0)
#  else
#    define WAIT_ASYNC() asm volatile("s_wait_asynccnt 0" ::: "memory")
#  endif
#endif

__device__ __forceinline__ float sigmf(float x) { return 1.0f / (1.0f + __expf(-x)); }

// WMMA 16x16x32 f16 per-lane fragment layouts (lm = lane&15, lh = lane>>4):
//   A: m = lm,  k = (i<8 ? i : i+8) + 8*lh  (+kbase)   -> two contiguous 8-half runs
//   B: k = (i<8 ? i : i+8) + 8*lh (+kbase), n = lm
//   C: m = r + 8*lh, n = lm
//
// Swizzled weight layout (both Wih and Whh), WSw[((nt*2 + kblk)*32 + lane)*16 + i]:
//   element = W[j = nt*16 + lm][k = kblk*32 + (i<8 ? i : i+8) + 8*lh]   (nt = N-tile of 16)
// -> a lane's B fragment is one contiguous 32-byte chunk.
//
// Swizzled Gin layout, Gin[TB*256 + lane*8 + r], TB = mtileGlobal*16 + g*4 + qt:
// -> a lane's C fragment is one contiguous 32-byte chunk.

// ---------------------------------------------------------------------------
// K0: weight swizzle+convert (f32 -> f16 fragment order) + bias sum
// ---------------------------------------------------------------------------
__global__ void prep_kernel(const float* __restrict__ Wih, const float* __restrict__ Whh,
                            const float* __restrict__ bih, const float* __restrict__ bhh,
                            _Float16* __restrict__ WihSw, _Float16* __restrict__ WhhSw,
                            float* __restrict__ bsum)
{
    const int tid = blockIdx.x * blockDim.x + threadIdx.x;   // 2048 threads
    if (tid < 1024) {                                        // 16 nt * 2 kblk * 32 lane
        const int lane = tid & 31;
        const int kblk = (tid >> 5) & 1;
        const int nt   = tid >> 6;
        const int lm = lane & 15, lh = lane >> 4;
        const int j  = nt * 16 + lm;
        const int kb = kblk * 32 + 8 * lh;
        v16h wi, wh;
        #pragma unroll
        for (int i = 0; i < 16; ++i) {
            wi[i] = (_Float16)Wih[j * DD + kb + ((i & 8) ? i + 8 : i)];
            wh[i] = (_Float16)Whh[j * QH + kb + ((i & 8) ? i + 8 : i)];
        }
        *(v16h*)(WihSw + tid * 16) = wi;
        *(v16h*)(WhhSw + tid * 16) = wh;
    }
    if (tid < G4) bsum[tid] = bih[tid] + bhh[tid];
}

// ---------------------------------------------------------------------------
// K1: parallel input-side GEMM for both LSTMs, output in fragment-swizzled Gin.
//     Y: m = t*64 + c (1024 M-tiles)   X: m = t*32 + n (512 M-tiles)
// Each wave: 1 M-tile x 4 N-tiles (one gate span).
// ---------------------------------------------------------------------------
__global__ void gin_kernel(const float* __restrict__ Yref, const float* __restrict__ traj,
                           const _Float16* __restrict__ WihSw, const float* __restrict__ bsum,
                           float* __restrict__ GinY, float* __restrict__ GinX)
{
    const int tid  = threadIdx.x;
    const int lane = tid & 31;
    const int wave = tid >> 5;
    const int lm   = lane & 15;
    const int lh   = lane >> 4;
    const int task = blockIdx.x * 8 + wave;            // 6144 wave tasks

    const bool isY = task < 4096;
    int t2 = isY ? task : task - 4096;
    if (!isY && t2 >= 2048) return;                    // wave-uniform exit
    const int mtile = t2 >> 2;                         // global M-tile index
    const int ngrp  = t2 & 3;
    const int Mbase = mtile * 16;
    const int mrow  = Mbase + lm;

    v16h a0, a1;
    if (isY) {
        // gathered column loads (one-shot, L2 resident)
        int c = mrow & 63, t = mrow >> 6;
        const float* src = Yref + (size_t)c * (DD * TT) + t;
        #pragma unroll
        for (int i = 0; i < 16; ++i) {
            int d0 = ((i & 8) ? (i + 8) : i) + 8 * lh;
            a0[i] = (_Float16)src[(size_t)d0 * TT];
            a1[i] = (_Float16)src[(size_t)(d0 + 32) * TT];
        }
    } else {
        int n = mrow & 31, t = mrow >> 5;
        const float* s0 = traj + (size_t)n * (TT * DD) + (size_t)t * DD + 8 * lh;
        const float4 f0 = *(const float4*)(s0);        // k = 8lh+0..3
        const float4 f1 = *(const float4*)(s0 + 4);    // k = 8lh+4..7
        const float4 f2 = *(const float4*)(s0 + 16);   // k = 16+8lh+0..3
        const float4 f3 = *(const float4*)(s0 + 20);
        const float4 f4 = *(const float4*)(s0 + 32);
        const float4 f5 = *(const float4*)(s0 + 36);
        const float4 f6 = *(const float4*)(s0 + 48);
        const float4 f7 = *(const float4*)(s0 + 52);
        a0[0]=(_Float16)f0.x; a0[1]=(_Float16)f0.y; a0[2]=(_Float16)f0.z; a0[3]=(_Float16)f0.w;
        a0[4]=(_Float16)f1.x; a0[5]=(_Float16)f1.y; a0[6]=(_Float16)f1.z; a0[7]=(_Float16)f1.w;
        a0[8]=(_Float16)f2.x; a0[9]=(_Float16)f2.y; a0[10]=(_Float16)f2.z; a0[11]=(_Float16)f2.w;
        a0[12]=(_Float16)f3.x; a0[13]=(_Float16)f3.y; a0[14]=(_Float16)f3.z; a0[15]=(_Float16)f3.w;
        a1[0]=(_Float16)f4.x; a1[1]=(_Float16)f4.y; a1[2]=(_Float16)f4.z; a1[3]=(_Float16)f4.w;
        a1[4]=(_Float16)f5.x; a1[5]=(_Float16)f5.y; a1[6]=(_Float16)f5.z; a1[7]=(_Float16)f5.w;
        a1[8]=(_Float16)f6.x; a1[9]=(_Float16)f6.y; a1[10]=(_Float16)f6.z; a1[11]=(_Float16)f6.w;
        a1[12]=(_Float16)f7.x; a1[13]=(_Float16)f7.y; a1[14]=(_Float16)f7.z; a1[15]=(_Float16)f7.w;
    }

    float* Gout = isY ? GinY : GinX;
    #pragma unroll
    for (int s = 0; s < 4; ++s) {
        const int nt = ngrp * 4 + s;                   // N-tile index 0..15
        const _Float16* bp = WihSw + (size_t)((nt * 2) * 32 + lane) * 16;
        const v16h b0 = *(const v16h*)bp;              // kblk 0
        const v16h b1 = *(const v16h*)(bp + 512);      // kblk 1
        v8f acc = {};
        acc = __builtin_amdgcn_wmma_f32_16x16x32_f16(false, a0, false, b0, (short)0, acc, false, false);
        acc = __builtin_amdgcn_wmma_f32_16x16x32_f16(false, a1, false, b1, (short)0, acc, false, false);
        const float bv = bsum[nt * 16 + lm];
        #pragma unroll
        for (int r = 0; r < 8; ++r) acc[r] += bv;
        const int TB = mtile * 16 + nt;
        *(v8f*)(Gout + (size_t)TB * 256 + lane * 8) = acc;   // fragment-order store
    }
}

// ---------------------------------------------------------------------------
// K2: recurrent LSTM. Grid = 2 blocks (0: Y batch 64, 1: X batch 32), 512 thr.
// Whh(f16 swizzled), h(f16), c(f32) resident in LDS (56 KB). Each wave owns one
// (m-tile, q-tile) and computes all four gate tiles -> gates fuse in registers.
// ---------------------------------------------------------------------------
__global__ void lstm_kernel(const float* __restrict__ h0Y, const float* __restrict__ c0Y,
                            const float* __restrict__ h0X, const float* __restrict__ c0X,
                            const _Float16* __restrict__ WhhSw,
                            const float* __restrict__ GinY, const float* __restrict__ GinX,
                            float* __restrict__ Ystore,   // [t][c][q]
                            float* __restrict__ Xstore)   // [n][t][q]
{
    __shared__ _Float16 whh[QH * G4];    // 32 KB, swizzled fragment order
    __shared__ _Float16 h16[64 * QH];    //  8 KB  [m][q]
    __shared__ float    cst[64 * QH];    // 16 KB  [m][q]

    const int tid  = threadIdx.x;        // 512
    const int lane = tid & 31;
    const int wave = tid >> 5;
    const int lm   = lane & 15;
    const int lh   = lane >> 4;

    const bool  isY = (blockIdx.x == 0);
    const int   B   = isY ? CTR : NN;
    const float* gin = isY ? GinY : GinX;
    const float* h0  = isY ? h0Y : h0X;
    const float* c0  = isY ? c0Y : c0X;

#ifdef HAS_ASYNC_LDS
    for (int i = tid; i < 2048; i += 512)            // 2048 x 16B = 32 KB
        async_b128((const char*)WhhSw + i * 16, (char*)whh + i * 16);
#else
    for (int i = tid; i < 2048; i += 512)
        *(float4*)((char*)whh + i * 16) = *(const float4*)((const char*)WhhSw + i * 16);
#endif
    for (int i = tid; i < B * QH; i += 512) { h16[i] = (_Float16)h0[i]; cst[i] = c0[i]; }
#ifdef HAS_ASYNC_LDS
    WAIT_ASYNC();
#endif
    __syncthreads();

    const int nTasks = (B / 16) * 4;     // 16 (Y) / 8 (X) wave tasks
    const bool active = wave < nTasks;
    const int mtile = wave >> 2;
    const int qtile = wave & 3;
    const int Mbase = mtile * 16;
    const int qb    = qtile * 16;

    for (int t = 0; t < TT; ++t) {
        v8f acc[4];
        if (active) {
            // A fragments: two contiguous 8-half runs per lane (ds_load_b128)
            const _Float16* hrow = h16 + (Mbase + lm) * QH + 8 * lh;
            const v8h a0l = *(const v8h*)(hrow);
            const v8h a0h = *(const v8h*)(hrow + 16);
            const v8h a1l = *(const v8h*)(hrow + 32);
            const v8h a1h = *(const v8h*)(hrow + 48);
            const v16h a0 = __builtin_shufflevector(a0l, a0h, 0,1,2,3,4,5,6,7,8,9,10,11,12,13,14,15);
            const v16h a1 = __builtin_shufflevector(a1l, a1h, 0,1,2,3,4,5,6,7,8,9,10,11,12,13,14,15);
            const int tb0 = (t * (B / 16) + mtile) * 16;
            #pragma unroll
            for (int g = 0; g < 4; ++g) {
                const int nt = g * 4 + qtile;
                const _Float16* bp = whh + (size_t)((nt * 2) * 32 + lane) * 16;
                const v16h b0 = *(const v16h*)bp;
                const v16h b1 = *(const v16h*)(bp + 512);
                v8f c = *(const v8f*)(gin + (size_t)(tb0 + nt) * 256 + lane * 8);
                c = __builtin_amdgcn_wmma_f32_16x16x32_f16(false, a0, false, b0, (short)0, c, false, false);
                c = __builtin_amdgcn_wmma_f32_16x16x32_f16(false, a1, false, b1, (short)0, c, false, false);
                acc[g] = c;
            }
        }
        __syncthreads();                 // all h16 reads complete before overwrite
        if (active) {
            #pragma unroll
            for (int r = 0; r < 8; ++r) {
                const int m = Mbase + r + 8 * lh;
                const int q = qb + lm;
                const float gi = acc[0][r], gf = acc[1][r], gg = acc[2][r], go = acc[3][r];
                const float cold = cst[m * QH + q];
                const float cn = sigmf(gf) * cold + sigmf(gi) * fast_tanhf(gg);
                const float hn = sigmf(go) * fast_tanhf(cn);
                cst[m * QH + q] = cn;
                h16[m * QH + q] = (_Float16)hn;
                if (isY) Ystore[(size_t)t * (CTR * QH) + m * QH + q] = hn;
                else     Xstore[(size_t)m * (TT * QH) + (size_t)t * QH + q] = hn;
            }
            if (t + 1 < TT) {            // global_prefetch_b8 of next Gin slab
                const float* pf = gin + (size_t)(t + 1) * B * G4 + tid * 32;
                __builtin_prefetch(pf, 0, 1);
            }
        }
        __syncthreads();
    }
}

// ---------------------------------------------------------------------------
// K3: gathered kernel distances + streamed softmax-weighted reduction.
// One block per (n,p); X_n (64 KB) LDS-resident via async b128; Y from L2.
// c = tid>>2, lg = tid&3; 4 l-streams reduced with wave32 shfl_xor.
// Writes sm[n][p][c] to its own slot -> fully deterministic (no atomics).
// ---------------------------------------------------------------------------
__global__ void phi_kernel(const float* __restrict__ Xstore, const float* __restrict__ Ystore,
                           const int* __restrict__ Aset, float* __restrict__ smbuf)
{
    __shared__ float Xl[TT * QH];        // 64 KB, [t][q]

    const int tid = threadIdx.x;         // 256
    const int n   = blockIdx.x >> 3;
    const int p   = blockIdx.x & 7;

    const char* gsrc = (const char*)(Xstore + (size_t)n * (TT * QH));
#ifdef HAS_ASYNC_LDS
    for (int i = tid; i < 4096; i += 256)               // 4096 x 16B = 64 KB
        async_b128(gsrc + i * 16, (char*)Xl + i * 16);
    WAIT_ASYNC();
#else
    for (int i = tid; i < 4096; i += 256)
        ((float4*)Xl)[i] = ((const float4*)gsrc)[i];
#endif
    __syncthreads();

    const int c  = tid >> 2;             // 0..63
    const int lg = tid & 3;              // 4 l-streams
    const int abase = ((n * PP + p) * LL) * 2;

    float S1 = 0.0f, S2 = 0.0f;
    for (int l = lg; l < LL; l += 4) {
        const int ia = Aset[abase + 2 * l];
        const int ib = Aset[abase + 2 * l + 1];
        const float4* yp = (const float4*)(Ystore + (size_t)ib * (CTR * QH) + c * QH);
        const float4* xp = (const float4*)(Xl + ia * QH);
        float dot = 0.0f, ysq = 0.0f, xsq = 0.0f;
        #pragma unroll
        for (int r = 0; r < QH / 4; ++r) {
            const float4 y = yp[r];
            const float4 x = xp[r];      // ds_load_b128, broadcast across c-lanes
            dot += y.x * x.x + y.y * x.y + y.z * x.z + y.w * x.w;
            ysq += y.x * y.x + y.y * y.y + y.z * y.z + y.w * y.w;
            xsq += x.x * x.x + x.y * x.y + x.z * x.z + x.w * x.w;
        }
        const float Dm  = xsq + ysq - 2.0f * dot;
        const float phi = __expf(-Dm * 0.1f);        // SCALE = 10
        const float e   = __expf(-phi);              // BETA = 1
        S1 += e;
        S2 += phi * e;
    }
    S1 += __shfl_xor(S1, 1, 32);  S1 += __shfl_xor(S1, 2, 32);
    S2 += __shfl_xor(S2, 1, 32);  S2 += __shfl_xor(S2, 2, 32);
    if (lg == 0) {
        smbuf[((size_t)n * PP + p) * CTR + c] = S2 / (512.0f * S1);
    }
}

// ---------------------------------------------------------------------------
// K4: odds = sum_p sm (fixed order) then FC (64x2) + SELU.
// One block, 64 threads: thread = (n, j).
// ---------------------------------------------------------------------------
__global__ void fc_kernel(const float* __restrict__ smbuf, const float* __restrict__ Wfc,
                          const float* __restrict__ bfc, float* __restrict__ out)
{
    const int tid = threadIdx.x;
    if (tid >= NN * 2) return;
    const int n = tid >> 1, j = tid & 1;
    float acc = bfc[j];
    const float* sb = smbuf + (size_t)n * PP * CTR;
    const float* w  = Wfc + j * CTR;
    for (int cc = 0; cc < CTR; ++cc) {
        float od = 0.0f;
        #pragma unroll
        for (int p = 0; p < PP; ++p) od += sb[p * CTR + cc];   // deterministic order
        acc += od * w[cc];
    }
    const float kScale = 1.0507009873554805f;
    const float kAlpha = 1.6732632423543772f;
    out[n * 2 + j] = acc > 0.0f ? kScale * acc : kScale * kAlpha * (__expf(acc) - 1.0f);
}

// ---------------------------------------------------------------------------
// Host launcher
// ---------------------------------------------------------------------------
extern "C" void kernel_launch(void* const* d_in, const int* in_sizes, int n_in,
                              void* d_out, int out_size, void* d_ws, size_t ws_size,
                              hipStream_t stream)
{
    (void)in_sizes; (void)n_in; (void)out_size; (void)ws_size;

    const float* Yref = (const float*)d_in[0];
    const float* traj = (const float*)d_in[1];
    const int*   Aset = (const int*)  d_in[2];
    const float* h0Y  = (const float*)d_in[3];
    const float* c0Y  = (const float*)d_in[4];
    const float* h0X  = (const float*)d_in[5];
    const float* c0X  = (const float*)d_in[6];
    const float* Wih  = (const float*)d_in[7];
    const float* Whh  = (const float*)d_in[8];
    const float* bih  = (const float*)d_in[9];
    const float* bhh  = (const float*)d_in[10];
    const float* Wfc  = (const float*)d_in[11];
    const float* bfc  = (const float*)d_in[12];
    float* out = (float*)d_out;

    // Workspace carve-up (~30 MB)
    char* ws = (char*)d_ws;
    _Float16* WihSw = (_Float16*)(ws + 0);                 // 32 KB
    _Float16* WhhSw = (_Float16*)(ws + 32768);             // 32 KB
    float*    bsum  = (float*)(ws + 65536);                // 1 KB
    float*    smbuf = (float*)(ws + 66560);                // N*P*CTR f32 = 64 KB
    float*    Ystore = (float*)(ws + 132096);              // 4 MB
    float*    Xstore = (float*)(ws + 132096 + 4194304);    // 2 MB
    float*    GinY   = (float*)(ws + 132096 + 4194304 + 2097152);  // 16 MB
    float*    GinX   = GinY + (size_t)TT * CTR * G4;               // 8 MB

    prep_kernel<<<dim3(8),       dim3(256), 0, stream>>>(Wih, Whh, bih, bhh, WihSw, WhhSw, bsum);
    gin_kernel <<<dim3(768),     dim3(256), 0, stream>>>(Yref, traj, WihSw, bsum, GinY, GinX);
    lstm_kernel<<<dim3(2),       dim3(512), 0, stream>>>(h0Y, c0Y, h0X, c0X, WhhSw, GinY, GinX, Ystore, Xstore);
    phi_kernel <<<dim3(NN * PP), dim3(256), 0, stream>>>(Xstore, Ystore, Aset, smbuf);
    fc_kernel  <<<dim3(1),       dim3(64),  0, stream>>>(smbuf, Wfc, bfc, out);
}
